// CausalSelfAttention_16020228014260
// MI455X (gfx1250) — compile-verified, matrix operand order checked by reading, not run on hardware
//
#include <hip/hip_runtime.h>
#include <hip/hip_bf16.h>

// ---------------------------------------------------------------------------
// Causal self-attention for MI455X (gfx1250, wave32, WMMA 16x16x32 f16).
// B=2, T=2048, C=1024, H=16, D=64. ~52 GFLOP, ~100MB HBM traffic -> matrix-
// core bound. All five GEMM-shaped ops run on v_wmma_f32_16x16x32_f16 with
// f32 accumulation; tiles are staged through LDS (double buffered, async
// GLOBAL_LOAD_ASYNC_TO_LDS_B128 when the toolchain exposes it).
// ---------------------------------------------------------------------------

typedef __attribute__((ext_vector_type(16))) _Float16 v16h;
typedef __attribute__((ext_vector_type(8)))  _Float16 v8h;
typedef __attribute__((ext_vector_type(8)))  float    v8f;
typedef __attribute__((ext_vector_type(4)))  int      v4i;

#define B_    2
#define T_    2048
#define C_    1024
#define H_    16
#define D_    64
#define MTOT  (B_ * T_)   // 4096

// ---- async global->LDS copy (CDNA5 ASYNCcnt path), with safe fallback ----
#if defined(__AMDGCN__) && __has_builtin(__builtin_amdgcn_global_load_async_to_lds_b128) && __has_builtin(__builtin_amdgcn_s_wait_asynccnt)
#define USE_ASYNC_LDS 1
#else
#define USE_ASYNC_LDS 0
#endif

#if USE_ASYNC_LDS
#define AS1__ __attribute__((address_space(1)))
#define AS3__ __attribute__((address_space(3)))
static __device__ __forceinline__ void copy_b128(const void* g, void* l) {
  // generic->AS1 is value preserving; AS3 offset = low 32 bits of generic LDS ptr
  __builtin_amdgcn_global_load_async_to_lds_b128(
      (AS1__ v4i*)(uintptr_t)g,
      (AS3__ v4i*)(uint32_t)(uintptr_t)l, 0, 0);
}
static __device__ __forceinline__ void wait_copies() {
  __builtin_amdgcn_s_wait_asynccnt(0);
}
#else
static __device__ __forceinline__ void copy_b128(const void* g, void* l) {
  *(v8h*)l = *(const v8h*)g;   // global_load_b128 -> ds_store_b128 fallback
}
static __device__ __forceinline__ void wait_copies() {}
#endif

static __device__ __forceinline__ v16h make_frag(v8h lo, v8h hi) {
  union { v16h v; v8h h[2]; } u;
  u.h[0] = lo; u.h[1] = hi;
  return u.v;
}

// Load one 16x32 (MxK) 16-bit WMMA operand fragment from row-major f16 data
// (global or LDS) with row stride `ld` elements. CDNA5 ISA layout:
//   lane<16 : row = lane,    K chunks {0..7} and {16..23}
//   lane>=16: row = lane-16, K chunks {8..15} and {24..31}
// B operands use the same pattern with "row" = N column (data stored N-major,
// which is why weights and V are pre-transposed).
static __device__ __forceinline__ v16h load_frag(const _Float16* src, int ld) {
  int lane = threadIdx.x & 31;
  int row  = lane & 15;
  int k0   = (lane >> 4) * 8;
  const v8h* p0 = (const v8h*)(src + (size_t)row * ld + k0);
  const v8h* p1 = (const v8h*)(src + (size_t)row * ld + k0 + 16);
  return make_frag(*p0, *p1);
}

static __device__ __forceinline__ v8f wmma16x16x32(v16h a, v16h b, v8f c) {
  return __builtin_amdgcn_wmma_f32_16x16x32_f16(
      /*neg_a=*/false, a, /*neg_b=*/false, b,
      /*c_mod=*/(short)0, c, /*reuse_a=*/false, /*reuse_b=*/false);
}

// ---------------------------------------------------------------------------
// elementwise conversion kernels
// ---------------------------------------------------------------------------
__global__ void cvt_f32_to_f16_kernel(const float* __restrict__ src,
                                      _Float16* __restrict__ dst, int n) {
  int i = blockIdx.x * 256 + threadIdx.x;
  if (i < n) dst[i] = (_Float16)src[i];
}

// src [rows][cols] f32 -> dst [cols][rows] f16  (weights become [N][K])
__global__ void transpose_to_f16_kernel(const float* __restrict__ src,
                                        _Float16* __restrict__ dst,
                                        int rows, int cols) {
  int i = blockIdx.x * 256 + threadIdx.x;
  if (i < rows * cols) {
    int r = i / cols, c = i % cols;
    dst[(size_t)c * rows + r] = (_Float16)src[i];
  }
}

// ---------------------------------------------------------------------------
// WMMA GEMM:  OUT[M][N] = A[M][K](f16) @ W (given as WT[N][K] f16) + bias
// 256 threads = 8 waves arranged 2(M) x 4(N); block tile 64 x 256; per-wave
// 32x64 = 8 accumulators; K step 32. Double-buffered LDS staging: tiles for
// step s+1 are loaded (async when available) while step s computes.
// mode 0/1: write f16 [b][h][t][d]   (Q / K)
// mode 2  : write f16 [b][h][d][t]   (V transposed, for contiguous PV frags)
// mode 3  : write f32 row-major + bias (final projection to d_out)
// ---------------------------------------------------------------------------
__global__ __launch_bounds__(256)
void gemm_wmma_kernel(const _Float16* __restrict__ A,
                      const _Float16* __restrict__ WT,
                      const float* __restrict__ bias,
                      _Float16* __restrict__ outH,
                      float* __restrict__ outF,
                      int Kdim, int Ndim, int mode) {
  __shared__ _Float16 As[2][64 * 32];    // 2 x 4 KB
  __shared__ _Float16 Bs[2][256 * 32];   // 2 x 16 KB

  int tid   = threadIdx.x;
  int lane  = tid & 31;
  int wave  = tid >> 5;
  int waveM = wave >> 2;                 // 0..1 -> 32 rows
  int waveN = wave & 3;                  // 0..3 -> 64 cols
  int mBase = blockIdx.x * 64;
  int nBase = blockIdx.y * 256;

  int arow = tid >> 2;                   // 0..63
  int acol = (tid & 3) * 8;              // f16 element offset within k-step

  auto stage = [&](int buf, int k0) {
    copy_b128(A + (size_t)(mBase + arow) * Kdim + k0 + acol,
              &As[buf][arow * 32 + acol]);
#pragma unroll
    for (int i = 0; i < 4; ++i) {
      int r = arow + i * 64;             // 0..255
      copy_b128(WT + (size_t)(nBase + r) * Kdim + k0 + acol,
                &Bs[buf][r * 32 + acol]);
    }
  };

  v8f acc[2][4] = {};
  int steps = Kdim / 32;

  stage(0, 0);
  wait_copies();
  __syncthreads();

  for (int s = 0; s < steps; ++s) {
    int cur = s & 1;
    if (s + 1 < steps) stage(cur ^ 1, (s + 1) * 32);

    v16h af[2], bf[4];
#pragma unroll
    for (int mf = 0; mf < 2; ++mf)
      af[mf] = load_frag(&As[cur][(waveM * 32 + mf * 16) * 32], 32);
#pragma unroll
    for (int j = 0; j < 4; ++j)
      bf[j] = load_frag(&Bs[cur][(waveN * 64 + j * 16) * 32], 32);
#pragma unroll
    for (int mf = 0; mf < 2; ++mf)
#pragma unroll
      for (int j = 0; j < 4; ++j)
        acc[mf][j] = wmma16x16x32(af[mf], bf[j], acc[mf][j]);

    if (s + 1 < steps) wait_copies();
    __syncthreads();
  }

  // C/D layout: VGPR r -> row = r + 8*(lane>=16), col = lane&15
  int col    = lane & 15;
  int rowOff = (lane >> 4) * 8;
#pragma unroll
  for (int mf = 0; mf < 2; ++mf) {
#pragma unroll
    for (int j = 0; j < 4; ++j) {
      int n = nBase + waveN * 64 + j * 16 + col;
      float bv = bias[n];
#pragma unroll
      for (int r = 0; r < 8; ++r) {
        int m = mBase + waveM * 32 + mf * 16 + rowOff + r;
        float val = acc[mf][j][r] + bv;
        if (mode == 3) {
          outF[(size_t)m * Ndim + n] = val;
        } else {
          int b = m / T_, t = m % T_;
          int h = n / D_, d = n % D_;
          if (mode == 2)
            outH[(((size_t)(b * H_ + h)) * D_ + d) * T_ + t] = (_Float16)val;
          else
            outH[(((size_t)(b * H_ + h)) * T_ + t) * D_ + d] = (_Float16)val;
        }
      }
    }
  }
}

// ---------------------------------------------------------------------------
// Flash attention, cooperative + double buffered.
// 256 threads = 8 waves; wave w owns query rows q0..q0+15. All waves share
// block-staged K (32x64) and V^T (64x32) tiles for the current 32-key block;
// per-wave causal clipping skips fully-masked blocks (barriers stay uniform).
// Per key block: 4 WMMAs for S = Q K^T, LDS round-trip converts C-layout
// probabilities to an A-layout fragment, 4 WMMAs for acc += P V.
// ---------------------------------------------------------------------------
__global__ __launch_bounds__(256)
void flash_attn_kernel(const _Float16* __restrict__ Qh,  // [B*H][T][D]
                       const _Float16* __restrict__ Kh,  // [B*H][T][D]
                       const _Float16* __restrict__ Vt,  // [B*H][D][T]
                       _Float16* __restrict__ Yh) {      // [B*T][C]
  __shared__ _Float16 Ks[2][32 * 64];    // 2 x 4 KB  (key-major)
  __shared__ _Float16 Vs[2][64 * 32];    // 2 x 4 KB  (d-major)
  __shared__ _Float16 Ps[8][16 * 32];    // 8 KB      (per-wave P tile)

  int tid  = threadIdx.x;
  int lane = tid & 31;
  int wave = tid >> 5;
  int h = blockIdx.y, b = blockIdx.z;
  int bh = b * H_ + h;
  int q0 = blockIdx.x * 128 + wave * 16;

  const _Float16* Kb = Kh + (size_t)bh * T_ * D_;
  const _Float16* Vb = Vt + (size_t)bh * D_ * T_;

  // cooperative staging assignment (one b128 of K + one of V per thread)
  int krow = tid >> 3, kcol = (tid & 7) * 8;   // K: 32 keys x 64 d
  int vrow = tid >> 2, vcol = (tid & 3) * 8;   // V: 64 d x 32 keys

  auto stage = [&](int buf, int s0) {
    copy_b128(Kb + (size_t)(s0 + krow) * D_ + kcol, &Ks[buf][krow * 64 + kcol]);
    copy_b128(Vb + (size_t)vrow * T_ + s0 + vcol,   &Vs[buf][vrow * 32 + vcol]);
  };

  const _Float16* Qb = Qh + ((size_t)bh * T_ + q0) * D_;
  v16h aQ0 = load_frag(Qb, D_);        // d = 0..31
  v16h aQ1 = load_frag(Qb + 32, D_);   // d = 32..63

  v8f acc[4] = {};
  float mrow[8], lrow[8];
#pragma unroll
  for (int r = 0; r < 8; ++r) { mrow[r] = -3.0e38f; lrow[r] = 0.0f; }

  int col    = lane & 15;
  int rowOff = (lane >> 4) * 8;
  _Float16* myP = Ps[wave];
  const float scale = 0.125f;          // 1/sqrt(64)

  int steps = blockIdx.x * 4 + 4;      // key blocks up to blockQ0+128
  stage(0, 0);
  wait_copies();
  __syncthreads();

  for (int s = 0; s < steps; ++s) {
    int cur = s & 1;
    int s0  = s * 32;
    if (s + 1 < steps) stage(cur ^ 1, (s + 1) * 32);

    if (s0 < q0 + 16) {                // wave-uniform causal clip
      // ---- scores: two 16x16 tiles over this 32-key block ----
      v8f c0 = {}, c1 = {};
      c0 = wmma16x16x32(aQ0, load_frag(&Ks[cur][0 * 64 + 0],  64), c0);
      c0 = wmma16x16x32(aQ1, load_frag(&Ks[cur][0 * 64 + 32], 64), c0);
      c1 = wmma16x16x32(aQ0, load_frag(&Ks[cur][16 * 64 + 0],  64), c1);
      c1 = wmma16x16x32(aQ1, load_frag(&Ks[cur][16 * 64 + 32], 64), c1);

      // ---- causal mask + scale, online softmax ----
      int key0 = s0 + col;
      int key1 = s0 + 16 + col;
      float p0[8], p1[8], rmax[8], rsum[8];
#pragma unroll
      for (int r = 0; r < 8; ++r) {
        int q = q0 + rowOff + r;
        p0[r] = (key0 <= q) ? c0[r] * scale : -3.0e38f;
        p1[r] = (key1 <= q) ? c1[r] * scale : -3.0e38f;
        rmax[r] = fmaxf(p0[r], p1[r]);
      }
      // a row spans 16 lanes of a wave32 half -> xor-shuffle reduce
#pragma unroll
      for (int off = 1; off < 16; off <<= 1)
#pragma unroll
        for (int r = 0; r < 8; ++r)
          rmax[r] = fmaxf(rmax[r], __shfl_xor(rmax[r], off, 32));

#pragma unroll
      for (int r = 0; r < 8; ++r) {
        float mnew  = fmaxf(mrow[r], rmax[r]);
        float alpha = __expf(mrow[r] - mnew);
        mrow[r] = mnew;
        p0[r] = __expf(p0[r] - mnew);
        p1[r] = __expf(p1[r] - mnew);
        rsum[r] = p0[r] + p1[r];
        lrow[r] *= alpha;
#pragma unroll
        for (int j = 0; j < 4; ++j) acc[j][r] *= alpha;
        int row = rowOff + r;
        myP[row * 32 + col]      = (_Float16)p0[r];
        myP[row * 32 + col + 16] = (_Float16)p1[r];
      }
#pragma unroll
      for (int off = 1; off < 16; off <<= 1)
#pragma unroll
        for (int r = 0; r < 8; ++r)
          rsum[r] += __shfl_xor(rsum[r], off, 32);
#pragma unroll
      for (int r = 0; r < 8; ++r) lrow[r] += rsum[r];

      // ---- acc += P @ V ---- (same-wave DS ops are in-order: no barrier)
      v16h aP = load_frag(myP, 32);
#pragma unroll
      for (int j = 0; j < 4; ++j)
        acc[j] = wmma16x16x32(aP, load_frag(&Vs[cur][(j * 16) * 32], 32), acc[j]);
    }

    if (s + 1 < steps) wait_copies();
    __syncthreads();
  }

  // ---- finalize: y = acc / l, merge heads back into [B*T][C] (f16) ----
#pragma unroll
  for (int r = 0; r < 8; ++r) {
    float inv = 1.0f / lrow[r];
    int t = q0 + rowOff + r;
#pragma unroll
    for (int j = 0; j < 4; ++j) {
      Yh[((size_t)(b * T_ + t)) * C_ + h * D_ + j * 16 + col] =
          (_Float16)(acc[j][r] * inv);
    }
  }
}

// ---------------------------------------------------------------------------
// launch
// ---------------------------------------------------------------------------
extern "C" void kernel_launch(void* const* d_in, const int* in_sizes, int n_in,
                              void* d_out, int out_size, void* d_ws, size_t ws_size,
                              hipStream_t stream) {
  const float* x  = (const float*)d_in[0];
  // d_in[1] = causal mask (computed analytically in-kernel)
  const float* Wq = (const float*)d_in[2];
  const float* bq = (const float*)d_in[3];
  const float* Wk = (const float*)d_in[4];
  const float* bk = (const float*)d_in[5];
  const float* Wv = (const float*)d_in[6];
  const float* bv = (const float*)d_in[7];
  const float* Wo = (const float*)d_in[8];
  const float* bo = (const float*)d_in[9];
  float* out = (float*)d_out;

  char* ws = (char*)d_ws;
  _Float16* xh  = (_Float16*)(ws);                        // 8 MB  [M][K]
  _Float16* WqT = (_Float16*)(ws + ((size_t)8  << 20));   // 2 MB  [N][K]
  _Float16* WkT = (_Float16*)(ws + ((size_t)10 << 20));
  _Float16* WvT = (_Float16*)(ws + ((size_t)12 << 20));
  _Float16* WoT = (_Float16*)(ws + ((size_t)14 << 20));
  _Float16* Qh  = (_Float16*)(ws + ((size_t)16 << 20));   // 8 MB [B*H][T][D]
  _Float16* Kh  = (_Float16*)(ws + ((size_t)24 << 20));   // 8 MB [B*H][T][D]
  _Float16* Vt  = (_Float16*)(ws + ((size_t)32 << 20));   // 8 MB [B*H][D][T]
  _Float16* Yh  = (_Float16*)(ws + ((size_t)40 << 20));   // 8 MB [B*T][C]

  int n = MTOT * C_;
  cvt_f32_to_f16_kernel<<<(n + 255) / 256, 256, 0, stream>>>(x, xh, n);
  n = C_ * C_;
  transpose_to_f16_kernel<<<(n + 255) / 256, 256, 0, stream>>>(Wq, WqT, C_, C_);
  transpose_to_f16_kernel<<<(n + 255) / 256, 256, 0, stream>>>(Wk, WkT, C_, C_);
  transpose_to_f16_kernel<<<(n + 255) / 256, 256, 0, stream>>>(Wv, WvT, C_, C_);
  transpose_to_f16_kernel<<<(n + 255) / 256, 256, 0, stream>>>(Wo, WoT, C_, C_);

  dim3 gg(MTOT / 64, C_ / 256);   // 64 x 4 blocks
  gemm_wmma_kernel<<<gg, 256, 0, stream>>>(xh, WqT, bq, Qh, nullptr, C_, C_, 0);
  gemm_wmma_kernel<<<gg, 256, 0, stream>>>(xh, WkT, bk, Kh, nullptr, C_, C_, 1);
  gemm_wmma_kernel<<<gg, 256, 0, stream>>>(xh, WvT, bv, Vt, nullptr, C_, C_, 2);

  dim3 gf(T_ / 128, H_, B_);      // 16 x 16 x 2 blocks
  flash_attn_kernel<<<gf, 256, 0, stream>>>(Qh, Kh, Vt, Yh);

  gemm_wmma_kernel<<<gg, 256, 0, stream>>>(Yh, WoT, bo, nullptr, out, C_, C_, 3);
}
